// CascadeModel_50199577755792
// MI455X (gfx1250) — matrix operand order repaired and checked by reading, not compiled, not run on hardware
//
#include <hip/hip_runtime.h>
#include <hip/hip_bf16.h>
#include <stddef.h>

// ---------------- model constants ----------------
#define B_  4
#define S_  512
#define H_  768
#define NH_ 12
#define HD_ 64
#define I_  3072
#define V_  50000
#define CE_ 256
#define NC_ 13
#define L_  12
#define LF_ 3

typedef __attribute__((ext_vector_type(16))) __bf16          v16bf;
typedef __attribute__((ext_vector_type(8)))  float           v8f;
typedef __attribute__((ext_vector_type(8)))  unsigned short  v8us;
typedef __attribute__((ext_vector_type(4)))  unsigned int    v4ui;
typedef __attribute__((ext_vector_type(8)))  int             v8i;
typedef __attribute__((ext_vector_type(4)))  int             v4i;

#if __has_builtin(__builtin_amdgcn_tensor_load_to_lds) && \
    __has_builtin(__builtin_amdgcn_s_wait_tensorcnt)
#define USE_TDM 1
#endif

union Frag16 {
    v16bf bf;
    v8us  us[2];
    unsigned short s[16];
};

__device__ __forceinline__ unsigned short f2bf(float f) {
    unsigned u = __float_as_uint(f);
    u += 0x7FFFu + ((u >> 16) & 1u);   // round-to-nearest-even
    return (unsigned short)(u >> 16);
}

// ---------------- generic batched bf16 WMMA GEMM ----------------
// Y[m,n] = act( alpha * sum_k A[m,k]*B(k,n) + bias[n] + resid[m,n] )
// bcol==0 -> B is W[N,K] row-major; bcol==1 -> B is [K,N] row-major.
struct GemmP {
    const unsigned short* A;
    const unsigned short* B;
    const float* bias;
    const float* resid;
    float* C;
    unsigned short* Cb;
    int M, N, K;
    int lda, ldb, ldc;
    long aso, asi, bso, bsi, cso, csi;  // batch offsets: z -> (z/bdiv)*so + (z%bdiv)*si
    int bdiv;
    float alpha;
    int act;    // 0 none, 1 relu, 2 exact gelu
    int bcol;
};

#ifdef USE_TDM
// Issue one 2D TDM tile load: tile_h rows of tile_w contiguous bf16 elements,
// source row stride = stride elements, zero-fill past (tens_w, tens_h).
// D# layout per CDNA5 ISA ch.8 (group0: count/lds/global/type, group1: dims).
__device__ __forceinline__ void tdm_load_2d(unsigned lds_addr, const void* gsrc,
                                            unsigned tile_w, unsigned tile_h,
                                            unsigned long tens_w, unsigned long tens_h,
                                            unsigned long stride) {
    const unsigned long ga = (unsigned long)gsrc;
    v4ui g0;
    g0[0] = 1u;                                          // count=1, user descriptor
    g0[1] = lds_addr;                                    // LDS byte address
    g0[2] = (unsigned)ga;                                // global_addr[31:0]
    g0[3] = (unsigned)((ga >> 32) & 0x1FFFFFFu) | 0x80000000u;  // addr[56:32] | type=2
    v8i g1;
    g1[0] = (int)0x00010000u;                            // data_size=1 (2 bytes)
    g1[1] = (int)((unsigned)(tens_w & 0xFFFFu) << 16);   // tensor_dim0[15:0]
    g1[2] = (int)(((tens_w >> 16) & 0xFFFFu) | ((tens_h & 0xFFFFu) << 16));
    g1[3] = (int)(((tens_h >> 16) & 0xFFFFu) | (tile_w << 16));   // tile_dim0
    g1[4] = (int)(tile_h & 0xFFFFu);                     // tile_dim1 (tile_dim2=0)
    g1[5] = (int)(stride & 0xFFFFFFFFu);                 // tensor_dim0_stride[31:0]
    g1[6] = (int)((stride >> 32) & 0xFFFFu);             // stride[47:32]
    g1[7] = 0;
    v4i gz = {0, 0, 0, 0};
#if __clang_major__ >= 23
    v8i gz8 = {0, 0, 0, 0, 0, 0, 0, 0};
    __builtin_amdgcn_tensor_load_to_lds(g0, g1, gz, gz, gz8, 0);
#else
    __builtin_amdgcn_tensor_load_to_lds(g0, g1, gz, gz, 0);
#endif
}
#endif

// Issue the A (64x32) and B (256x32 or 32x256) tile loads for step k0.
__device__ __forceinline__ void stage_issue(const GemmP& p,
                                            unsigned short* Asb, unsigned short* Bsb,
                                            long aoff, long boff,
                                            int m0blk, int n0blk, int k0) {
#ifdef USE_TDM
    if (threadIdx.x == 0) {
        tdm_load_2d((unsigned)(size_t)Asb,
                    p.A + aoff + (long)m0blk * p.lda + k0, 32, 64,
                    (unsigned long)(p.K - k0), (unsigned long)(p.M - m0blk),
                    (unsigned long)p.lda);
        const unsigned long nrem = (p.N > n0blk) ? (unsigned long)(p.N - n0blk) : 0ul;
        if (!p.bcol)
            tdm_load_2d((unsigned)(size_t)Bsb,
                        p.B + boff + (long)n0blk * p.ldb + k0, 32, 256,
                        (unsigned long)(p.K - k0), nrem, (unsigned long)p.ldb);
        else
            tdm_load_2d((unsigned)(size_t)Bsb,
                        p.B + boff + (long)k0 * p.ldb + n0blk, 256, 32,
                        nrem, (unsigned long)(p.K - k0), (unsigned long)p.ldb);
    }
#else
    // Fallback: cooperative staged copy (zero-fill OOB columns).
    const int t = threadIdx.x;
    for (int i = t; i < 64 * 32; i += 256) {
        const int r = i >> 5, c = i & 31;
        Asb[i] = p.A[aoff + (long)(m0blk + r) * p.lda + k0 + c];
    }
    if (!p.bcol) {
        for (int i = t; i < 256 * 32; i += 256) {
            const int r = i >> 5, c = i & 31;
            const int n = n0blk + r;
            Bsb[i] = (n < p.N) ? p.B[boff + (long)n * p.ldb + k0 + c]
                               : (unsigned short)0;
        }
    } else {
        for (int i = t; i < 32 * 256; i += 256) {
            const int r = i >> 8, c = i & 255;
            const int n = n0blk + c;
            Bsb[i] = (n < p.N) ? p.B[boff + (long)(k0 + r) * p.ldb + n]
                               : (unsigned short)0;
        }
    }
#endif
}

__device__ __forceinline__ void stage_wait() {
#ifdef USE_TDM
    if (threadIdx.x == 0) __builtin_amdgcn_s_wait_tensorcnt(0);
#endif
    __syncthreads();
}

__global__ void __launch_bounds__(256, 1) gemm_bf16_wmma(GemmP p) {
    __shared__ unsigned short As[2][64 * 32];     //  8 KB: [m][k]
    __shared__ unsigned short Bs[2][256 * 32];    // 32 KB: W-form [n][k] / bcol [k][n]

    const int lane = threadIdx.x & 31;
    const int wid  = threadIdx.x >> 5;
    const int wm   = wid >> 2;          // 0..1
    const int wn   = wid & 3;           // 0..3
    const int half = lane >> 4;         // 0..1
    const int l15  = lane & 15;

    const int z  = blockIdx.z;
    const int zo = z / p.bdiv;
    const int zi = z - zo * p.bdiv;
    const long aoff = (long)zo * p.aso + (long)zi * p.asi;
    const long boff = (long)zo * p.bso + (long)zi * p.bsi;
    const long coff = (long)zo * p.cso + (long)zi * p.csi;

    const int m0blk = blockIdx.x * 64;
    const int n0blk = blockIdx.y * 256;

    v8f acc[2][4] = {};

    stage_issue(p, As[0], Bs[0], aoff, boff, m0blk, n0blk, 0);
    stage_wait();

    const int nsteps = p.K >> 5;
    for (int s = 0; s < nsteps; ++s) {
        const int buf = s & 1;
        if (s + 1 < nsteps)
            stage_issue(p, As[buf ^ 1], Bs[buf ^ 1], aoff, boff,
                        m0blk, n0blk, (s + 1) << 5);

        Frag16 a[2], b[4];
        #pragma unroll
        for (int mi = 0; mi < 2; ++mi) {
            const unsigned short* pa =
                &As[buf][(wm * 32 + mi * 16 + l15) * 32 + half * 16];
            a[mi].us[0] = *(const v8us*)(pa);
            a[mi].us[1] = *(const v8us*)(pa + 8);
        }
        if (!p.bcol) {
            #pragma unroll
            for (int ni = 0; ni < 4; ++ni) {
                const unsigned short* pb =
                    &Bs[buf][(wn * 64 + ni * 16 + l15) * 32 + half * 16];
                b[ni].us[0] = *(const v8us*)(pb);
                b[ni].us[1] = *(const v8us*)(pb + 8);
            }
        } else {
            #pragma unroll
            for (int ni = 0; ni < 4; ++ni) {
                #pragma unroll
                for (int j = 0; j < 16; ++j)
                    b[ni].s[j] =
                        Bs[buf][(half * 16 + j) * 256 + wn * 64 + ni * 16 + l15];
            }
        }
        #pragma unroll
        for (int mi = 0; mi < 2; ++mi)
            #pragma unroll
            for (int ni = 0; ni < 4; ++ni)
                acc[mi][ni] = __builtin_amdgcn_wmma_f32_16x16x32_bf16(
                    false, a[mi].bf, false, b[ni].bf,
                    (short)0, acc[mi][ni], false, false);

        stage_wait();
    }

    // epilogue: D element (m,n): vgpr r -> m = r + 8*half within tile; lane l15 -> n
    #pragma unroll
    for (int mi = 0; mi < 2; ++mi) {
        const int mb = m0blk + wm * 32 + mi * 16 + half * 8;
        #pragma unroll
        for (int ni = 0; ni < 4; ++ni) {
            const int n = n0blk + wn * 64 + ni * 16 + l15;
            if (n >= p.N) continue;
            const float bsv = p.bias ? p.bias[n] : 0.f;
            #pragma unroll
            for (int r = 0; r < 8; ++r) {
                const long off = coff + (long)(mb + r) * p.ldc + n;
                float v = acc[mi][ni][r] * p.alpha + bsv;
                if (p.resid) v += p.resid[off];
                if (p.act == 1)      v = v > 0.f ? v : 0.f;
                else if (p.act == 2) v = 0.5f * v * (1.f + erff(v * 0.70710678118654752f));
                if (p.C)  p.C[off]  = v;
                if (p.Cb) p.Cb[off] = f2bf(v);
            }
        }
    }
}

// ---------------- elementwise / reduction kernels ----------------
__global__ void convert_kernel(const float* __restrict__ src,
                               unsigned short* __restrict__ dst, size_t n) {
    size_t i = (size_t)blockIdx.x * blockDim.x + threadIdx.x;
    const size_t stride = (size_t)gridDim.x * blockDim.x;
    for (; i < n; i += stride) dst[i] = f2bf(src[i]);
}

__global__ void nodeflat_kernel(const int* __restrict__ nids,
                                const float* __restrict__ cw,
                                const float* __restrict__ nemb,
                                const float* __restrict__ wpw,
                                const float* __restrict__ wpb,
                                unsigned short* __restrict__ out) {
    const long i = (long)blockIdx.x * 256 + threadIdx.x;
    if (i >= (long)S_ * NC_ * CE_) return;
    const int  e  = (int)(i % CE_);
    const long sc = i / CE_;                 // s*NC + c
    const float v = nemb[(long)nids[sc] * CE_ + e] + cw[sc] * wpw[e] + wpb[e];
    out[i] = f2bf(v);
}

__global__ void embed_kernel(const int* __restrict__ ids,
                             const float* __restrict__ tok,
                             const float* __restrict__ pos,
                             const float* __restrict__ casch,
                             float* __restrict__ out) {
    const long i = (long)blockIdx.x * 256 + threadIdx.x;
    if (i >= (long)B_ * S_ * H_) return;
    const int  h  = (int)(i % H_);
    const long bs = i / H_;
    const int  s  = (int)(bs % S_);
    out[i] = tok[(long)ids[bs] * H_ + h] + pos[(long)s * H_ + h] + casch[(long)s * H_ + h];
}

// LayerNorm over width 768; writes fp32 and bf16 copies.
__global__ void __launch_bounds__(256) ln_kernel(const float* __restrict__ X,
                                                 const float* __restrict__ g,
                                                 const float* __restrict__ b,
                                                 float* __restrict__ Y,
                                                 unsigned short* __restrict__ Yb,
                                                 float eps) {
    __shared__ float red[256];
    const long row = blockIdx.x;
    const int  t   = threadIdx.x;
    const float* x = X + row * H_;
    const float v0 = x[t], v1 = x[t + 256], v2 = x[t + 512];
    red[t] = v0 + v1 + v2;
    __syncthreads();
    for (int st = 128; st > 0; st >>= 1) {
        if (t < st) red[t] += red[t + st];
        __syncthreads();
    }
    const float mean = red[0] * (1.f / 768.f);
    __syncthreads();
    const float d0 = v0 - mean, d1 = v1 - mean, d2 = v2 - mean;
    red[t] = d0 * d0 + d1 * d1 + d2 * d2;
    __syncthreads();
    for (int st = 128; st > 0; st >>= 1) {
        if (t < st) red[t] += red[t + st];
        __syncthreads();
    }
    const float inv = rsqrtf(red[0] * (1.f / 768.f) + eps);
    float* y = Y + row * H_;
    unsigned short* yb = Yb + row * H_;
    const float o0 = d0 * inv * g[t]       + b[t];
    const float o1 = d1 * inv * g[t + 256] + b[t + 256];
    const float o2 = d2 * inv * g[t + 512] + b[t + 512];
    y[t] = o0; y[t + 256] = o1; y[t + 512] = o2;
    yb[t] = f2bf(o0); yb[t + 256] = f2bf(o1); yb[t + 512] = f2bf(o2);
}

// softmax over 512 keys; QK already *1/sqrt(d); CB already *CW (may be null). P is bf16.
__global__ void __launch_bounds__(256) softmax_kernel(const float* __restrict__ QK,
                                                      const float* __restrict__ CB,
                                                      unsigned short* __restrict__ P,
                                                      int nh) {
    __shared__ float red[256];
    const int t = threadIdx.x;
    const int q = blockIdx.x;
    const int z = blockIdx.y;           // b*NH + h
    const int bb = z / nh;
    const float* row = QK + ((long)z * S_ + q) * S_;
    float a0 = row[t], a1 = row[t + 256];
    if (CB) {
        const float* crow = CB + ((long)bb * S_ + q) * S_;
        a0 += crow[t]; a1 += crow[t + 256];
    }
    red[t] = fmaxf(a0, a1);
    __syncthreads();
    for (int st = 128; st > 0; st >>= 1) {
        if (t < st) red[t] = fmaxf(red[t], red[t + st]);
        __syncthreads();
    }
    const float m = red[0];
    __syncthreads();
    const float e0 = expf(a0 - m), e1 = expf(a1 - m);
    red[t] = e0 + e1;
    __syncthreads();
    for (int st = 128; st > 0; st >>= 1) {
        if (t < st) red[t] += red[t + st];
        __syncthreads();
    }
    const float inv = 1.f / red[0];
    unsigned short* prow = P + ((long)z * S_ + q) * S_;
    prow[t] = f2bf(e0 * inv);
    prow[t + 256] = f2bf(e1 * inv);
}

// ---------------- host orchestration ----------------
extern "C" void kernel_launch(void* const* d_in, const int* in_sizes, int n_in,
                              void* d_out, int out_size, void* d_ws, size_t ws_size,
                              hipStream_t stream) {
    (void)in_sizes; (void)n_in; (void)out_size; (void)ws_size;

    const int*   ids    = (const int*)d_in[0];
    const int*   nids   = (const int*)d_in[1];
    const float* cw     = (const float*)d_in[2];
    const float* cc     = (const float*)d_in[3];
    const float* tok    = (const float*)d_in[4];
    const float* pos    = (const float*)d_in[5];
    const float* nemb   = (const float*)d_in[6];
    const float* wpw    = (const float*)d_in[7];
    const float* wpb    = (const float*)d_in[8];
    const float* fuw    = (const float*)d_in[9];
    const float* fub    = (const float*)d_in[10];
    const float* elng   = (const float*)d_in[11];
    const float* elnb   = (const float*)d_in[12];
    const float* w_fqkv = (const float*)d_in[13];
    const float* b_fqkv = (const float*)d_in[14];
    const float* w_fout = (const float*)d_in[15];
    const float* b_fout = (const float*)d_in[16];
    const float* w_fff1 = (const float*)d_in[17];
    const float* b_fff1 = (const float*)d_in[18];
    const float* w_fff2 = (const float*)d_in[19];
    const float* b_fff2 = (const float*)d_in[20];
    const float* g_fln1 = (const float*)d_in[21];
    const float* bt_fln1= (const float*)d_in[22];
    const float* g_fln2 = (const float*)d_in[23];
    const float* bt_fln2= (const float*)d_in[24];
    const float* w_q    = (const float*)d_in[25];
    const float* b_q    = (const float*)d_in[26];
    const float* w_k    = (const float*)d_in[27];
    const float* b_k    = (const float*)d_in[28];
    const float* w_v    = (const float*)d_in[29];
    const float* b_v    = (const float*)d_in[30];
    const float* w_cq   = (const float*)d_in[31];
    const float* b_cq   = (const float*)d_in[32];
    const float* w_ck   = (const float*)d_in[33];
    const float* b_ck   = (const float*)d_in[34];
    const float* w_o    = (const float*)d_in[35];
    const float* b_o    = (const float*)d_in[36];
    const float* w_ff1  = (const float*)d_in[37];
    const float* b_ff1  = (const float*)d_in[38];
    const float* w_ff2  = (const float*)d_in[39];
    const float* b_ff2  = (const float*)d_in[40];
    const float* g_ln1  = (const float*)d_in[41];
    const float* bt_ln1 = (const float*)d_in[42];
    const float* g_ln2  = (const float*)d_in[43];
    const float* bt_ln2 = (const float*)d_in[44];
    const float* g_oln  = (const float*)d_in[45];
    const float* bt_oln = (const float*)d_in[46];
    const float* w_lm   = (const float*)d_in[47];

    char* base = (char*)d_ws;
    size_t off = 0;
    auto alloc = [&](size_t bytes) -> void* {
        void* r = base + off;
        off += (bytes + 255) & ~(size_t)255;
        return r;
    };

    auto cvt = [&](const float* src, unsigned short* dst, size_t n) {
        size_t blocks = (n + 255) / 256;
        if (blocks > 32768) blocks = 32768;
        convert_kernel<<<dim3((unsigned)blocks), 256, 0, stream>>>(src, dst, n);
    };

    auto gemm = [&](const unsigned short* A, const unsigned short* Bm, const float* bias,
                    const float* resid, float* C, unsigned short* Cb,
                    int M, int N, int K, int lda, int ldb, int ldc,
                    long aso, long asi, long bso, long bsi, long cso, long csi,
                    int bdiv, int batches, float alpha, int act, int bcol) {
        GemmP p{A, Bm, bias, resid, C, Cb, M, N, K, lda, ldb, ldc,
                aso, asi, bso, bsi, cso, csi, bdiv, alpha, act, bcol};
        dim3 grid(M / 64, (N + 255) / 256, batches);
        gemm_bf16_wmma<<<grid, 256, 0, stream>>>(p);
    };

    // ---- bf16 weight caches ----
    unsigned short* wb_cascfu = (unsigned short*)alloc((size_t)H_ * NC_ * CE_ * 2);
    unsigned short *wb_fqkv[LF_], *wb_fout[LF_], *wb_fff1[LF_], *wb_fff2[LF_];
    for (int lf = 0; lf < LF_; ++lf) {
        wb_fqkv[lf] = (unsigned short*)alloc((size_t)3 * H_ * H_ * 2);
        wb_fout[lf] = (unsigned short*)alloc((size_t)H_ * H_ * 2);
        wb_fff1[lf] = (unsigned short*)alloc((size_t)I_ * H_ * 2);
        wb_fff2[lf] = (unsigned short*)alloc((size_t)H_ * I_ * 2);
    }
    unsigned short *wb_q[L_], *wb_k[L_], *wb_v[L_], *wb_cqw[L_], *wb_ckw[L_],
                   *wb_o[L_], *wb_f1[L_], *wb_f2[L_];
    for (int l = 0; l < L_; ++l) {
        wb_q[l]   = (unsigned short*)alloc((size_t)H_ * H_ * 2);
        wb_k[l]   = (unsigned short*)alloc((size_t)H_ * H_ * 2);
        wb_v[l]   = (unsigned short*)alloc((size_t)H_ * H_ * 2);
        wb_cqw[l] = (unsigned short*)alloc((size_t)H_ * CE_ * 2);
        wb_ckw[l] = (unsigned short*)alloc((size_t)H_ * CE_ * 2);
        wb_o[l]   = (unsigned short*)alloc((size_t)H_ * H_ * 2);
        wb_f1[l]  = (unsigned short*)alloc((size_t)I_ * H_ * 2);
        wb_f2[l]  = (unsigned short*)alloc((size_t)H_ * I_ * 2);
    }
    unsigned short* wb_lm = (unsigned short*)alloc((size_t)V_ * H_ * 2);

    // ---- activations ----
    unsigned short* cc_b   = (unsigned short*)alloc((size_t)B_ * S_ * CE_ * 2);
    unsigned short* ndflat = (unsigned short*)alloc((size_t)S_ * NC_ * CE_ * 2);
    float*          casch  = (float*)alloc((size_t)S_ * H_ * 4);
    float*          x      = (float*)alloc((size_t)B_ * S_ * H_ * 4);
    unsigned short* x_b    = (unsigned short*)alloc((size_t)B_ * S_ * H_ * 2);
    float*          t1     = (float*)alloc((size_t)B_ * S_ * H_ * 4);
    unsigned short* t0_b   = (unsigned short*)alloc((size_t)B_ * S_ * I_ * 2);
    unsigned short* qbuf   = (unsigned short*)alloc((size_t)B_ * S_ * H_ * 2);
    unsigned short* kbuf   = (unsigned short*)alloc((size_t)B_ * S_ * H_ * 2);
    unsigned short* vbuf   = (unsigned short*)alloc((size_t)B_ * S_ * H_ * 2);
    unsigned short* cqbuf  = (unsigned short*)alloc((size_t)B_ * S_ * H_ * 2);
    unsigned short* ckbuf  = (unsigned short*)alloc((size_t)B_ * S_ * H_ * 2);
    unsigned short* ctxb   = (unsigned short*)alloc((size_t)B_ * S_ * H_ * 2);
    float*          qk     = (float*)alloc((size_t)B_ * NH_ * S_ * S_ * 4);
    float*          cbuf   = (float*)alloc((size_t)B_ * S_ * S_ * 4);
    unsigned short* Pb     = (unsigned short*)alloc((size_t)B_ * NH_ * S_ * S_ * 2);

    // ---- weight conversions ----
    cvt(fuw, wb_cascfu, (size_t)H_ * NC_ * CE_);
    for (int lf = 0; lf < LF_; ++lf) {
        cvt(w_fqkv + (size_t)lf * 3 * H_ * H_, wb_fqkv[lf], (size_t)3 * H_ * H_);
        cvt(w_fout + (size_t)lf * H_ * H_,     wb_fout[lf], (size_t)H_ * H_);
        cvt(w_fff1 + (size_t)lf * I_ * H_,     wb_fff1[lf], (size_t)I_ * H_);
        cvt(w_fff2 + (size_t)lf * H_ * I_,     wb_fff2[lf], (size_t)H_ * I_);
    }
    for (int l = 0; l < L_; ++l) {
        cvt(w_q  + (size_t)l * H_ * H_,  wb_q[l],   (size_t)H_ * H_);
        cvt(w_k  + (size_t)l * H_ * H_,  wb_k[l],   (size_t)H_ * H_);
        cvt(w_v  + (size_t)l * H_ * H_,  wb_v[l],   (size_t)H_ * H_);
        cvt(w_cq + (size_t)l * H_ * CE_, wb_cqw[l], (size_t)H_ * CE_);
        cvt(w_ck + (size_t)l * H_ * CE_, wb_ckw[l], (size_t)H_ * CE_);
        cvt(w_o  + (size_t)l * H_ * H_,  wb_o[l],   (size_t)H_ * H_);
        cvt(w_ff1+ (size_t)l * I_ * H_,  wb_f1[l],  (size_t)I_ * H_);
        cvt(w_ff2+ (size_t)l * H_ * I_,  wb_f2[l],  (size_t)H_ * I_);
    }
    cvt(w_lm, wb_lm, (size_t)V_ * H_);
    cvt(cc, cc_b, (size_t)B_ * S_ * CE_);

    const float inv = 0.125f;   // 1/sqrt(64)
    const float CW  = 0.3f;

    // ---- cascade embedding ----
    {
        const long n = (long)S_ * NC_ * CE_;
        nodeflat_kernel<<<dim3((unsigned)((n + 255) / 256)), 256, 0, stream>>>(
            nids, cw, nemb, wpw, wpb, ndflat);
    }
    gemm(ndflat, wb_cascfu, fub, nullptr, casch, nullptr,
         S_, H_, NC_ * CE_, NC_ * CE_, NC_ * CE_, H_,
         0, 0, 0, 0, 0, 0, 1, 1, 1.f, 0, 0);
    {
        const long n = (long)B_ * S_ * H_;
        embed_kernel<<<dim3((unsigned)((n + 255) / 256)), 256, 0, stream>>>(
            ids, tok, pos, casch, t1);
    }
    ln_kernel<<<B_ * S_, 256, 0, stream>>>(t1, elng, elnb, x, x_b, 1e-12f);

    // ---- fusion layers (post-norm, relu FFN, eps 1e-5) ----
    for (int lf = 0; lf < LF_; ++lf) {
        gemm(x_b, wb_fqkv[lf], b_fqkv + (size_t)lf * 3 * H_, nullptr, nullptr, t0_b,
             B_ * S_, 3 * H_, H_, H_, H_, 3 * H_,
             0, 0, 0, 0, 0, 0, 1, 1, 1.f, 0, 0);
        gemm(t0_b, t0_b + H_, nullptr, nullptr, qk, nullptr,
             S_, S_, HD_, 3 * H_, 3 * H_, S_,
             (long)S_ * 3 * H_, HD_, (long)S_ * 3 * H_, HD_,
             (long)NH_ * S_ * S_, (long)S_ * S_,
             NH_, B_ * NH_, inv, 0, 0);
        softmax_kernel<<<dim3(S_, B_ * NH_), 256, 0, stream>>>(qk, nullptr, Pb, NH_);
        gemm(Pb, t0_b + 2 * H_, nullptr, nullptr, nullptr, ctxb,
             S_, HD_, S_, S_, 3 * H_, H_,
             (long)NH_ * S_ * S_, (long)S_ * S_, (long)S_ * 3 * H_, HD_,
             (long)S_ * H_, HD_,
             NH_, B_ * NH_, 1.f, 0, 1);
        gemm(ctxb, wb_fout[lf], b_fout + (size_t)lf * H_, x, t1, nullptr,
             B_ * S_, H_, H_, H_, H_, H_,
             0, 0, 0, 0, 0, 0, 1, 1, 1.f, 0, 0);
        ln_kernel<<<B_ * S_, 256, 0, stream>>>(t1, g_fln1 + (size_t)lf * H_,
                                               bt_fln1 + (size_t)lf * H_, x, x_b, 1e-5f);
        gemm(x_b, wb_fff1[lf], b_fff1 + (size_t)lf * I_, nullptr, nullptr, t0_b,
             B_ * S_, I_, H_, H_, H_, I_,
             0, 0, 0, 0, 0, 0, 1, 1, 1.f, 1 /*relu*/, 0);
        gemm(t0_b, wb_fff2[lf], b_fff2 + (size_t)lf * H_, x, t1, nullptr,
             B_ * S_, H_, I_, I_, I_, H_,
             0, 0, 0, 0, 0, 0, 1, 1, 1.f, 0, 0);
        ln_kernel<<<B_ * S_, 256, 0, stream>>>(t1, g_fln2 + (size_t)lf * H_,
                                               bt_fln2 + (size_t)lf * H_, x, x_b, 1e-5f);
    }

    // ---- main cascade layers (eps 1e-12, exact gelu FFN) ----
    for (int l = 0; l < L_; ++l) {
        gemm(x_b, wb_q[l], b_q + (size_t)l * H_, nullptr, nullptr, qbuf,
             B_ * S_, H_, H_, H_, H_, H_, 0, 0, 0, 0, 0, 0, 1, 1, 1.f, 0, 0);
        gemm(x_b, wb_k[l], b_k + (size_t)l * H_, nullptr, nullptr, kbuf,
             B_ * S_, H_, H_, H_, H_, H_, 0, 0, 0, 0, 0, 0, 1, 1, 1.f, 0, 0);
        gemm(x_b, wb_v[l], b_v + (size_t)l * H_, nullptr, nullptr, vbuf,
             B_ * S_, H_, H_, H_, H_, H_, 0, 0, 0, 0, 0, 0, 1, 1, 1.f, 0, 0);
        gemm(cc_b, wb_cqw[l], b_cq + (size_t)l * H_, nullptr, nullptr, cqbuf,
             B_ * S_, H_, CE_, CE_, CE_, H_, 0, 0, 0, 0, 0, 0, 1, 1, 1.f, 0, 0);
        gemm(cc_b, wb_ckw[l], b_ck + (size_t)l * H_, nullptr, nullptr, ckbuf,
             B_ * S_, H_, CE_, CE_, CE_, H_, 0, 0, 0, 0, 0, 0, 1, 1, 1.f, 0, 0);
        // scores = inv * Q K^T  (batched over b,h)
        gemm(qbuf, kbuf, nullptr, nullptr, qk, nullptr,
             S_, S_, HD_, H_, H_, S_,
             (long)S_ * H_, HD_, (long)S_ * H_, HD_,
             (long)NH_ * S_ * S_, (long)S_ * S_,
             NH_, B_ * NH_, inv, 0, 0);
        // cascade bias = CW * cq ck^T (batched over b)
        gemm(cqbuf, ckbuf, nullptr, nullptr, cbuf, nullptr,
             S_, S_, H_, H_, H_, S_,
             (long)S_ * H_, 0, (long)S_ * H_, 0, (long)S_ * S_, 0,
             1, B_, CW, 0, 0);
        softmax_kernel<<<dim3(S_, B_ * NH_), 256, 0, stream>>>(qk, cbuf, Pb, NH_);
        gemm(Pb, vbuf, nullptr, nullptr, nullptr, ctxb,
             S_, HD_, S_, S_, H_, H_,
             (long)NH_ * S_ * S_, (long)S_ * S_, (long)S_ * H_, HD_,
             (long)S_ * H_, HD_,
             NH_, B_ * NH_, 1.f, 0, 1);
        gemm(ctxb, wb_o[l], b_o + (size_t)l * H_, x, t1, nullptr,
             B_ * S_, H_, H_, H_, H_, H_, 0, 0, 0, 0, 0, 0, 1, 1, 1.f, 0, 0);
        ln_kernel<<<B_ * S_, 256, 0, stream>>>(t1, g_ln1 + (size_t)l * H_,
                                               bt_ln1 + (size_t)l * H_, x, x_b, 1e-12f);
        gemm(x_b, wb_f1[l], b_ff1 + (size_t)l * I_, nullptr, nullptr, t0_b,
             B_ * S_, I_, H_, H_, H_, I_, 0, 0, 0, 0, 0, 0, 1, 1, 1.f, 2 /*gelu*/, 0);
        gemm(t0_b, wb_f2[l], b_ff2 + (size_t)l * H_, x, t1, nullptr,
             B_ * S_, H_, I_, I_, I_, H_, 0, 0, 0, 0, 0, 0, 1, 1, 1.f, 0, 0);
        ln_kernel<<<B_ * S_, 256, 0, stream>>>(t1, g_ln2 + (size_t)l * H_,
                                               bt_ln2 + (size_t)l * H_, x, x_b, 1e-12f);
    }

    // ---- output LN + LM head ----
    ln_kernel<<<B_ * S_, 256, 0, stream>>>(x, g_oln, bt_oln, t1, x_b, 1e-12f);
    gemm(x_b, wb_lm, nullptr, nullptr, (float*)d_out, nullptr,
         B_ * S_, V_, H_, H_, H_, V_,
         0, 0, 0, 0, 0, 0, 1, 1, 1.f, 0, 0);
}
